// MultiScaleHistoryPooler_3685081940188
// MI455X (gfx1250) — compile-verified, matrix operand order checked by reading, not run on hardware
//
#include <hip/hip_runtime.h>
#include <hip/hip_bf16.h>
#include <cstdint>
#include <cstddef>

// ---------------------------------------------------------------------------
// MultiScaleHistoryPooler for MI455X (gfx1250), wave32 + bf16 WMMA + TDM.
// B=8 L=4096 D=512 P=128 scales={1,5,32,128} O=512
// ---------------------------------------------------------------------------

#define B_      8
#define L_      4096
#define D_      512
#define P_      128
#define NS_     4
#define O_      512
#define ROWS_   (B_ * L_)          // 32768
#define CHUNK_  512
#define NCHUNK_ (L_ / CHUNK_)      // 8
#define LDS_ROW_BYTES 80           // 32 bf16 (64B) + 16B pad (bank spread)
#define TILE_BYTES (P_ * LDS_ROW_BYTES)   // 10240 B per K-chunk tile

typedef __attribute__((ext_vector_type(16))) __bf16 v16bf;
typedef __attribute__((ext_vector_type(8)))  float  v8f;

union BfFrag { v16bf v; uint4 u[2]; };

__device__ __forceinline__ int scale_of(int s) {
    return (s == 0) ? 1 : (s == 1) ? 5 : (s == 2) ? 32 : 128;
}

// ---------------------------------------------------------------------------
// Tensor Data Mover staging of one 128-row x 32-bf16-col weight tile
// (row stride 512 bf16 in memory) into LDS with hardware padding to an
// 80-byte row stride. Issued once per wave (EXEC ignored by TDM).
// ---------------------------------------------------------------------------
#if defined(__has_builtin)
#if __has_builtin(__builtin_amdgcn_tensor_load_to_lds) && \
    __has_builtin(__builtin_amdgcn_s_wait_tensorcnt)
#define USE_TDM 1
#endif
#endif

#if defined(USE_TDM)
typedef unsigned int u32x4 __attribute__((ext_vector_type(4)));
typedef int          i32x4 __attribute__((ext_vector_type(4)));
typedef int          i32x8 __attribute__((ext_vector_type(8)));

__device__ __forceinline__ void tdm_stage_tile(const __bf16* gsrc, unsigned ldsOff) {
    unsigned long long ga = (unsigned long long)(uintptr_t)gsrc;
    u32x4 g0;
    g0[0] = 1u;                                    // count = 1 (valid D#)
    g0[1] = ldsOff;                                // lds_addr (bytes)
    g0[2] = (unsigned)(ga & 0xffffffffull);        // global_addr[31:0]
    g0[3] = (unsigned)((ga >> 32) & 0x01ffffffull) // global_addr[56:32]
          | (2u << 30);                            // type = 2 ("image")
    i32x8 g1;
    g1[0] = (1 << 16)    // data_size = 1 -> 2 bytes (bf16)
          | (1 << 20)    // pad_enable
          | (3 << 22)    // pad_interval = 16 DWORDs (64B row)
          | (3 << 25);   // pad_amount   = 4 DWORDs (16B) -> 80B LDS row stride
    g1[1] = (D_ << 16);  // tensor_dim0 = 512 (bits 79:48, low half)
    g1[2] = (P_ << 16);  // tensor_dim1 = 128 (bits 111:80, low half)
    g1[3] = (32 << 16);  // tile_dim0 = 32 elements (bits 127:112)
    g1[4] = P_;          // tile_dim1 = 128 rows (bits 143:128); tile_dim2 = 0
    g1[5] = D_;          // tensor_dim0_stride = 512 elements (bits 207:160, low)
    g1[6] = 0;           // stride0 hi, stride1 lo
    g1[7] = 0;           // stride1 hi
    i32x4 z4 = {0, 0, 0, 0};
#if __clang_major__ >= 23
    i32x8 z8 = {0, 0, 0, 0, 0, 0, 0, 0};
    __builtin_amdgcn_tensor_load_to_lds(g0, g1, z4, z4, z8, 0);
#else
    __builtin_amdgcn_tensor_load_to_lds(g0, g1, z4, z4, 0);
#endif
}
#endif // USE_TDM

// Fallback: cooperative synchronous staging (256 threads, 2 x uint4 each).
__device__ __forceinline__ void stage_sync(const __bf16* gsrc,
                                           unsigned char* dst, int tid) {
#pragma unroll
    for (int it = 0; it < 2; ++it) {
        int i  = tid + it * 256;          // 0..511
        int nr = i >> 2;
        int g  = i & 3;
        *(uint4*)(dst + nr * LDS_ROW_BYTES + g * 16) =
            *(const uint4*)(gsrc + (size_t)nr * 512 + g * 8);
    }
}

// ---------------------------------------------------------------------------
// Kernel 0: f32 -> bf16 weight conversion
// ---------------------------------------------------------------------------
__global__ void k_convert_bf16(const float* __restrict__ src,
                               __bf16* __restrict__ dst, int n) {
    int i = blockIdx.x * blockDim.x + threadIdx.x;
    if (i < n) dst[i] = (__bf16)src[i];
}

// ---------------------------------------------------------------------------
// Kernel 1: chunked local cumsum along L. Thread = (b, chunk, d).
// ---------------------------------------------------------------------------
__global__ void k_cumsum_local(const float* __restrict__ x,
                               float* __restrict__ cs,
                               float* __restrict__ T) {
    int tid = blockIdx.x * blockDim.x + threadIdx.x;   // 0..32767
    int d   = tid & (D_ - 1);
    int bc  = tid >> 9;                                // b*NCHUNK + c
    int b   = bc >> 3;
    int c   = bc & (NCHUNK_ - 1);
    size_t base = ((size_t)b * L_ + (size_t)c * CHUNK_) * D_ + d;
    float acc = 0.f;
    for (int j = 0; j < CHUNK_; ++j) {
        acc += x[base + (size_t)j * D_];
        cs[base + (size_t)j * D_] = acc;
    }
    T[(size_t)bc * D_ + d] = acc;
}

// ---------------------------------------------------------------------------
// Kernel 2: fixup — add exclusive prefix of chunk totals.
// ---------------------------------------------------------------------------
__global__ void k_cumsum_fixup(float* __restrict__ cs,
                               const float* __restrict__ T) {
    int tid = blockIdx.x * blockDim.x + threadIdx.x;
    int d   = tid & (D_ - 1);
    int bc  = tid >> 9;
    int c   = bc & (NCHUNK_ - 1);
    if (c == 0) return;
    int b = bc >> 3;
    float off = 0.f;
    for (int cc = 0; cc < c; ++cc)
        off += T[((size_t)(b * NCHUNK_ + cc)) * D_ + d];
    size_t base = ((size_t)b * L_ + (size_t)c * CHUNK_) * D_ + d;
    for (int j = 0; j < CHUNK_; ++j)
        cs[base + (size_t)j * D_] += off;
}

// ---------------------------------------------------------------------------
// Shared GEMM double-buffer staging step (TDM async or sync fallback).
// ---------------------------------------------------------------------------
__device__ __forceinline__ void stage_next_and_sync(const __bf16* nextSrc,
                                                    unsigned char* nxt,
                                                    bool haveNext,
                                                    int tid, int wave) {
#if defined(USE_TDM)
    if (haveNext && wave == 0)
        tdm_stage_tile(nextSrc, (unsigned)(uintptr_t)nxt);
    if (haveNext) __builtin_amdgcn_s_wait_tensorcnt(1);   // current tile done
    else          __builtin_amdgcn_s_wait_tensorcnt(0);
    __syncthreads();
#else
    (void)wave;
    if (haveNext) stage_sync(nextSrc, nxt, tid);
    __syncthreads();
#endif
}

// ---------------------------------------------------------------------------
// Kernel 3: fused pooled-GEMM + exact GELU per scale.
//   grid = (ROWS/128, NS), block = 256 (8 waves).
// ---------------------------------------------------------------------------
__global__ __launch_bounds__(256)
void k_scale_gemm(const float* __restrict__ cs,
                  const __bf16* __restrict__ Wsb,
                  const float* __restrict__ bs,
                  __bf16* __restrict__ cat) {
    __shared__ __align__(16) unsigned char lds[2 * TILE_BYTES];

    const int tid  = threadIdx.x;
    const int lane = tid & 31;
    const int wave = tid >> 5;
    const int half = lane >> 4;
    const int mrow = lane & 15;

    const int s = blockIdx.y;
    const int n = scale_of(s);
    const int rowBase = blockIdx.x * 128 + wave * 16;

    const int pA = rowBase + mrow;
    const int l  = pA & (L_ - 1);
    const float invd = 1.0f / (float)((l + 1 < n) ? (l + 1) : n);
    const size_t rowHi = (size_t)pA * D_;
    const bool  hasLo  = (l >= n);
    const size_t rowLo = rowHi - (size_t)n * D_;

    const __bf16* Wbase = Wsb + (size_t)s * P_ * D_;

    v8f acc[8];
#pragma unroll
    for (int t = 0; t < 8; ++t) {
#pragma unroll
        for (int v = 0; v < 8; ++v) acc[t][v] = 0.f;
    }

    // prologue: stage first tile into buffer 0
#if defined(USE_TDM)
    if (wave == 0) tdm_stage_tile(Wbase, (unsigned)(uintptr_t)&lds[0]);
#else
    stage_sync(Wbase, &lds[0], tid);
#endif

    for (int kb = 0; kb < D_; kb += 32) {
        const int curIdx = (kb >> 5) & 1;
        unsigned char* cur = lds + curIdx * TILE_BYTES;
        unsigned char* nxt = lds + (curIdx ^ 1) * TILE_BYTES;
        const bool haveNext = (kb + 32) < D_;

        stage_next_and_sync(Wbase + kb + 32, nxt, haveNext, tid, wave);

        // ---- build A fragment (pooled, bf16) ----
        const int kA0 = kb + half * 8;
        const int kA1 = kb + 16 + half * 8;
        __align__(16) float hi[16], lo[16];
        *(float4*)&hi[0]  = *(const float4*)(cs + rowHi + kA0);
        *(float4*)&hi[4]  = *(const float4*)(cs + rowHi + kA0 + 4);
        *(float4*)&hi[8]  = *(const float4*)(cs + rowHi + kA1);
        *(float4*)&hi[12] = *(const float4*)(cs + rowHi + kA1 + 4);
        if (hasLo) {
            *(float4*)&lo[0]  = *(const float4*)(cs + rowLo + kA0);
            *(float4*)&lo[4]  = *(const float4*)(cs + rowLo + kA0 + 4);
            *(float4*)&lo[8]  = *(const float4*)(cs + rowLo + kA1);
            *(float4*)&lo[12] = *(const float4*)(cs + rowLo + kA1 + 4);
        } else {
#pragma unroll
            for (int e = 0; e < 16; ++e) lo[e] = 0.f;
        }
        v16bf a;
#pragma unroll
        for (int e = 0; e < 16; ++e)
            a[e] = (__bf16)((hi[e] - lo[e]) * invd);

        // ---- 8 WMMA column tiles from current LDS buffer ----
#pragma unroll
        for (int t = 0; t < 8; ++t) {
            const unsigned char* bp =
                cur + (t * 16 + mrow) * LDS_ROW_BYTES + half * 16;
            BfFrag bf;
            bf.u[0] = *(const uint4*)(bp);
            bf.u[1] = *(const uint4*)(bp + 32);
            acc[t] = __builtin_amdgcn_wmma_f32_16x16x32_bf16(
                false, a, false, bf.v, (short)0, acc[t], false, false);
        }
        __syncthreads();   // protect cur before it is re-staged next iteration
    }

    // ---- epilogue: bias + exact GELU -> bf16 cat ----
#pragma unroll
    for (int t = 0; t < 8; ++t) {
        const int ncol = t * 16 + mrow;
        const float bias = bs[s * P_ + ncol];
#pragma unroll
        for (int v = 0; v < 8; ++v) {
            const int m = v + half * 8;
            const int p = rowBase + m;
            float x = acc[t][v] + bias;
            float g = 0.5f * x * (1.0f + erff(x * 0.70710678118654752f));
            cat[(size_t)p * O_ + s * P_ + ncol] = (__bf16)g;
        }
    }
}

// ---------------------------------------------------------------------------
// Kernel 4: combine GEMM. out[p, o] = cat[p,:] @ Wc[o,:]^T + bc[o]
//   grid = (ROWS/128, O/128), block = 256 (8 waves).
// ---------------------------------------------------------------------------
__global__ __launch_bounds__(256)
void k_combine_gemm(const __bf16* __restrict__ cat,
                    const __bf16* __restrict__ Wcb,
                    const float* __restrict__ bc,
                    float* __restrict__ out) {
    __shared__ __align__(16) unsigned char lds[2 * TILE_BYTES];

    const int tid  = threadIdx.x;
    const int lane = tid & 31;
    const int wave = tid >> 5;
    const int half = lane >> 4;
    const int mrow = lane & 15;

    const int rowBase = blockIdx.x * 128 + wave * 16;
    const int nbase   = blockIdx.y * 128;

    const int pA = rowBase + mrow;
    const __bf16* arow = cat + (size_t)pA * O_;
    const __bf16* Wbase = Wcb + (size_t)nbase * O_;

    v8f acc[8];
#pragma unroll
    for (int t = 0; t < 8; ++t) {
#pragma unroll
        for (int v = 0; v < 8; ++v) acc[t][v] = 0.f;
    }

#if defined(USE_TDM)
    if (wave == 0) tdm_stage_tile(Wbase, (unsigned)(uintptr_t)&lds[0]);
#else
    stage_sync(Wbase, &lds[0], tid);
#endif

    for (int kb = 0; kb < O_; kb += 32) {
        const int curIdx = (kb >> 5) & 1;
        unsigned char* cur = lds + curIdx * TILE_BYTES;
        unsigned char* nxt = lds + (curIdx ^ 1) * TILE_BYTES;
        const bool haveNext = (kb + 32) < O_;

        stage_next_and_sync(Wbase + kb + 32, nxt, haveNext, tid, wave);

        // A fragment straight from bf16 cat
        BfFrag af;
        af.u[0] = *(const uint4*)(arow + kb + half * 8);
        af.u[1] = *(const uint4*)(arow + kb + 16 + half * 8);

#pragma unroll
        for (int t = 0; t < 8; ++t) {
            const unsigned char* bp =
                cur + (t * 16 + mrow) * LDS_ROW_BYTES + half * 16;
            BfFrag bf;
            bf.u[0] = *(const uint4*)(bp);
            bf.u[1] = *(const uint4*)(bp + 32);
            acc[t] = __builtin_amdgcn_wmma_f32_16x16x32_bf16(
                false, af.v, false, bf.v, (short)0, acc[t], false, false);
        }
        __syncthreads();
    }

#pragma unroll
    for (int t = 0; t < 8; ++t) {
        const int ncol = nbase + t * 16 + mrow;
        const float bias = bc[ncol];
#pragma unroll
        for (int v = 0; v < 8; ++v) {
            const int m = v + half * 8;
            const int p = rowBase + m;
            out[(size_t)p * O_ + ncol] = acc[t][v] + bias;
        }
    }
}

// ---------------------------------------------------------------------------
// Launch
// ---------------------------------------------------------------------------
extern "C" void kernel_launch(void* const* d_in, const int* in_sizes, int n_in,
                              void* d_out, int out_size, void* d_ws, size_t ws_size,
                              hipStream_t stream) {
    (void)in_sizes; (void)n_in; (void)out_size; (void)ws_size;
    const float* emb = (const float*)d_in[0];   // (8,4096,512)
    const float* Ws  = (const float*)d_in[1];   // (4,128,512)
    const float* bs  = (const float*)d_in[2];   // (4,128)
    const float* Wc  = (const float*)d_in[3];   // (512,512)
    const float* bc  = (const float*)d_in[4];   // (512,)
    float* out = (float*)d_out;                 // (8,4096,512)

    char* ws = (char*)d_ws;
    const size_t csBytes  = (size_t)ROWS_ * D_ * sizeof(float);        // 64 MB
    const size_t tBytes   = (size_t)B_ * NCHUNK_ * D_ * sizeof(float); // 128 KB
    const size_t catBytes = (size_t)ROWS_ * O_ * sizeof(__bf16);       // 32 MB
    const size_t wsbBytes = (size_t)NS_ * P_ * D_ * sizeof(__bf16);    // 512 KB

    float*  CS  = (float*)ws;
    float*  T   = (float*)(ws + csBytes);
    __bf16* cat = (__bf16*)(ws + csBytes + tBytes);
    __bf16* Wsb = (__bf16*)(ws + csBytes + tBytes + catBytes);
    __bf16* Wcb = (__bf16*)(ws + csBytes + tBytes + catBytes + wsbBytes);

    // weights -> bf16
    k_convert_bf16<<<(NS_ * P_ * D_ + 255) / 256, 256, 0, stream>>>(Ws, Wsb, NS_ * P_ * D_);
    k_convert_bf16<<<(O_ * O_ + 255) / 256, 256, 0, stream>>>(Wc, Wcb, O_ * O_);

    // cumsum (2-pass parallel scan)
    k_cumsum_local<<<(B_ * NCHUNK_ * D_) / 256, 256, 0, stream>>>(emb, CS, T);
    k_cumsum_fixup<<<(B_ * NCHUNK_ * D_) / 256, 256, 0, stream>>>(CS, T);

    // fused pooling + per-scale GEMM + GELU (CS stays hot in 192MB L2)
    dim3 g2(ROWS_ / 128, NS_);
    k_scale_gemm<<<g2, 256, 0, stream>>>(CS, Wsb, bs, cat);

    // combine GEMM
    dim3 g3(ROWS_ / 128, O_ / 128);
    k_combine_gemm<<<g3, 256, 0, stream>>>(cat, Wcb, bc, out);
}
